// MultiHeadSelfAttention_37254546326204
// MI455X (gfx1250) — matrix operand order repaired and checked by reading, not compiled, not run on hardware
//
#include <hip/hip_runtime.h>
#include <hip/hip_bf16.h>

typedef __attribute__((ext_vector_type(16))) __bf16 v16bf;
typedef __attribute__((ext_vector_type(8)))  float  v8f;
typedef __attribute__((ext_vector_type(4)))  int    v4i;

struct alignas(16) U4 { unsigned int x, y, z, w; };

union FragA {
    v16bf  v;
    __bf16 h[16];
    U4     q[2];
};

#if __has_builtin(__builtin_amdgcn_global_load_async_to_lds_b128)
#define USE_ASYNC_LDS 1
#define ASYNC_GPTR(p) ((__attribute__((address_space(1))) v4i*)(p))
#define ASYNC_LPTR(p) ((__attribute__((address_space(3))) v4i*)(p))
#else
#define USE_ASYNC_LDS 0
#endif

// ---------------------------------------------------------------------------
// f32 -> bf16 conversion (4 elements / thread)
// ---------------------------------------------------------------------------
__global__ __launch_bounds__(256)
void f32_to_bf16_kernel(const float* __restrict__ src, __bf16* __restrict__ dst, int n) {
    int i4 = (blockIdx.x * blockDim.x + threadIdx.x) * 4;
    if (i4 + 3 < n) {
        float4 f = *reinterpret_cast<const float4*>(src + i4);
        dst[i4 + 0] = (__bf16)f.x;
        dst[i4 + 1] = (__bf16)f.y;
        dst[i4 + 2] = (__bf16)f.z;
        dst[i4 + 3] = (__bf16)f.w;
    }
}

// ---------------------------------------------------------------------------
// f32 (R x C) -> bf16 transposed (C x R), LDS-tiled for coalescing.
// ---------------------------------------------------------------------------
__global__ __launch_bounds__(256)
void f32_to_bf16_transpose_kernel(const float* __restrict__ src, __bf16* __restrict__ dst,
                                  int R, int C) {
    __shared__ float tile[32][33];
    const int r0 = blockIdx.y * 32, c0 = blockIdx.x * 32;
    const int tx = threadIdx.x & 31, ty = threadIdx.x >> 5;
#pragma unroll
    for (int i = 0; i < 32; i += 8)
        tile[ty + i][tx] = src[(size_t)(r0 + ty + i) * C + c0 + tx];
    __syncthreads();
#pragma unroll
    for (int i = 0; i < 32; i += 8)
        dst[(size_t)(c0 + ty + i) * R + r0 + tx] = (__bf16)tile[tx][ty + i];
}

// ---------------------------------------------------------------------------
// Tiled bf16 GEMM + bias: C(M,N) = A(M,K) @ W(K,N) + bias(N)
// Wt is PRE-TRANSPOSED (N x K) so B-fragments are contiguous 32B loads.
// out_mode 0: f32 row-major (M,N)
// out_mode 1: bf16 head-major (B,H,S,Dk)            [Q, K]
// out_mode 2: bf16 head-major transposed (B,H,Dk,S) [V]
// Block tile 128x128x32, 256 threads = 8 waves, wave tile 64x32 (4x2 WMMA).
// Async path: GLOBAL_LOAD_ASYNC_TO_LDS_B128 with double-buffered LDS,
// one barrier per k-tile; loads for tile i+1 overlap WMMAs of tile i.
// ---------------------------------------------------------------------------
__global__ __launch_bounds__(256)
void gemm_bias_bf16_kernel(const __bf16* __restrict__ A,
                           const __bf16* __restrict__ Wt,
                           const float*  __restrict__ bias,
                           void* __restrict__ out,
                           int M, int N, int K, int out_mode) {
#if USE_ASYNC_LDS
    __shared__ __bf16 As[2][128][32 + 24];    // row stride 112B, conflict-free
    __shared__ __bf16 Bsn[2][128][32 + 24];
#else
    __shared__ __bf16 As[1][128][32 + 24];
    __shared__ __bf16 Bsn[1][128][32 + 24];
#endif

    const int tid  = threadIdx.x;
    const int wid  = tid >> 5;
    const int lane = tid & 31;
    const int lr   = lane & 15;
    const int hi   = lane >> 4;
    const int klo  = hi * 8;

    const int bm = blockIdx.y * 128;
    const int bn = blockIdx.x * 128;
    const int wm0 = (wid & 1) * 64;
    const int wn0 = (wid >> 1) * 32;

    v8f acc[4][2];
#pragma unroll
    for (int mi = 0; mi < 4; ++mi)
#pragma unroll
        for (int ni = 0; ni < 2; ++ni)
#pragma unroll
            for (int r = 0; r < 8; ++r) acc[mi][ni][r] = 0.0f;

    const int row = tid >> 1, kc = (tid & 1) * 16;   // 128 rows x 32 k-cols
    const __bf16* gA = A  + (size_t)(bm + row) * K + kc;
    const __bf16* gB = Wt + (size_t)(bn + row) * K + kc;

#if USE_ASYNC_LDS
    auto issue_tile = [&](int buf, int k0) {
        const __bf16* ga = gA + k0;
        const __bf16* gb = gB + k0;
        __builtin_amdgcn_global_load_async_to_lds_b128(
            ASYNC_GPTR(ga),     ASYNC_LPTR(&As[buf][row][kc]),      0, 0);
        __builtin_amdgcn_global_load_async_to_lds_b128(
            ASYNC_GPTR(ga + 8), ASYNC_LPTR(&As[buf][row][kc + 8]),  0, 0);
        __builtin_amdgcn_global_load_async_to_lds_b128(
            ASYNC_GPTR(gb),     ASYNC_LPTR(&Bsn[buf][row][kc]),     0, 0);
        __builtin_amdgcn_global_load_async_to_lds_b128(
            ASYNC_GPTR(gb + 8), ASYNC_LPTR(&Bsn[buf][row][kc + 8]), 0, 0);
    };

    const int nk = K >> 5;
    issue_tile(0, 0);
    for (int i = 0; i < nk; ++i) {
        asm volatile("s_wait_asynccnt 0x0" ::: "memory");   // own tile-i loads done
        __syncthreads();                                    // all waves' tile i in LDS
        if (i + 1 < nk) issue_tile((i + 1) & 1, (i + 1) << 5);
        const int buf = i & 1;
#else
    for (int k0 = 0; k0 < K; k0 += 32) {
        const U4* ga = reinterpret_cast<const U4*>(gA + k0);
        U4 a0 = ga[0], a1 = ga[1];
        const U4* gb = reinterpret_cast<const U4*>(gB + k0);
        U4 b0 = gb[0], b1 = gb[1];
        if (k0 + 32 < K) {   // gfx1250 global_prefetch of next k-tile
            __builtin_prefetch(gA + k0 + 32, 0, 3);
            __builtin_prefetch(gB + k0 + 32, 0, 3);
        }
        __syncthreads();
        *reinterpret_cast<U4*>(&As[0][row][kc])      = a0;
        *reinterpret_cast<U4*>(&As[0][row][kc + 8])  = a1;
        *reinterpret_cast<U4*>(&Bsn[0][row][kc])     = b0;
        *reinterpret_cast<U4*>(&Bsn[0][row][kc + 8]) = b1;
        __syncthreads();
        const int buf = 0;
#endif

        FragA afrag[4];
#pragma unroll
        for (int mi = 0; mi < 4; ++mi) {
            const __bf16* ap = &As[buf][wm0 + mi * 16 + lr][0];
            afrag[mi].q[0] = *reinterpret_cast<const U4*>(ap + hi * 16);
            afrag[mi].q[1] = *reinterpret_cast<const U4*>(ap + hi * 16 + 8);
        }
        FragA bfrag[2];
#pragma unroll
        for (int ni = 0; ni < 2; ++ni) {
            const __bf16* bp = &Bsn[buf][wn0 + ni * 16 + lr][hi * 16];
            bfrag[ni].q[0] = *reinterpret_cast<const U4*>(bp);
            bfrag[ni].q[1] = *reinterpret_cast<const U4*>(bp + 8);
        }
#pragma unroll
        for (int mi = 0; mi < 4; ++mi)
#pragma unroll
            for (int ni = 0; ni < 2; ++ni)
                acc[mi][ni] = __builtin_amdgcn_wmma_f32_16x16x32_bf16(
                    false, afrag[mi].v, false, bfrag[ni].v,
                    (short)0, acc[mi][ni], false, false);
    }

#pragma unroll
    for (int mi = 0; mi < 4; ++mi) {
#pragma unroll
        for (int ni = 0; ni < 2; ++ni) {
            const int gcol = bn + wn0 + ni * 16 + lr;
            const float bv = bias[gcol];
#pragma unroll
            for (int r = 0; r < 8; ++r) {
                const int grow = bm + wm0 + mi * 16 + r + hi * 8;
                const float val = acc[mi][ni][r] + bv;
                if (out_mode == 0) {
                    reinterpret_cast<float*>(out)[(size_t)grow * N + gcol] = val;
                } else {
                    const int b = grow >> 11, s = grow & 2047;
                    const int h = gcol >> 6,  d = gcol & 63;
                    if (out_mode == 1) {   // (B,H,S,Dk)
                        reinterpret_cast<__bf16*>(out)
                            [(((size_t)(b * 16 + h) * 2048 + s) << 6) + d] = (__bf16)val;
                    } else {               // (B,H,Dk,S)
                        reinterpret_cast<__bf16*>(out)
                            [(((size_t)(b * 16 + h) * 64 + d) << 11) + s] = (__bf16)val;
                    }
                }
            }
        }
    }
}

// ---------------------------------------------------------------------------
// Causal flash attention, bf16 WMMA, f32 online softmax.
// Q/K: (B,H,S,Dk) bf16.  V: (B,H,Dk,S) bf16 (pre-transposed).
// ctx out: (B,S,H*Dk) bf16.
// ---------------------------------------------------------------------------
__global__ __launch_bounds__(128)
void attn_causal_kernel(const __bf16* __restrict__ Q,
                        const __bf16* __restrict__ Km,
                        const __bf16* __restrict__ V,
                        __bf16* __restrict__ ctx) {
    __shared__ __bf16 Plds[4][16][32];   // per-wave P staging (C-layout -> A-layout)

    const int tid  = threadIdx.x;
    const int wid  = tid >> 5;
    const int lane = tid & 31;
    const int lr   = lane & 15;
    const int hi   = lane >> 4;
    const int klo  = hi * 8;

    const int bh = blockIdx.y;          // b*16 + h
    const int b  = bh >> 4, h = bh & 15;
    const int qw = blockIdx.x * 64 + wid * 16;

    const __bf16* qbase = Q  + ((size_t)bh * 2048 + qw) * 64;
    const __bf16* kbase = Km + (size_t)bh * 2048 * 64;
    const __bf16* vbase = V  + (size_t)bh * 64 * 2048;

    FragA qf[2];
#pragma unroll
    for (int c = 0; c < 2; ++c) {
        const __bf16* qp = qbase + (size_t)lr * 64 + c * 32;
        qf[c].q[0] = *reinterpret_cast<const U4*>(qp + hi * 16);
        qf[c].q[1] = *reinterpret_cast<const U4*>(qp + hi * 16 + 8);
    }

    float mrow[8], lrow[8];
    v8f accO[4];
#pragma unroll
    for (int r = 0; r < 8; ++r) { mrow[r] = -__builtin_inff(); lrow[r] = 0.0f; }
#pragma unroll
    for (int dc = 0; dc < 4; ++dc)
#pragma unroll
        for (int r = 0; r < 8; ++r) accO[dc][r] = 0.0f;

    const float scale = 0.125f;          // 1/sqrt(64)
    const int   kv_end = qw + 16;

    for (int kv = 0; kv < kv_end; kv += 32) {
        if (kv + 32 < kv_end) {          // gfx1250 prefetch of next K tile rows
            __builtin_prefetch(kbase + (size_t)(kv + 32 + lane) * 64, 0, 3);
        }

        // ---- scores S = Q @ K^T (16 x 32) ----
        v8f sc[2];
#pragma unroll
        for (int ns = 0; ns < 2; ++ns) {
#pragma unroll
            for (int r = 0; r < 8; ++r) sc[ns][r] = 0.0f;
            const __bf16* kp = kbase + (size_t)(kv + ns * 16 + lr) * 64;
#pragma unroll
            for (int c = 0; c < 2; ++c) {
                FragA kf;
                kf.q[0] = *reinterpret_cast<const U4*>(kp + c * 32 + hi * 16);
                kf.q[1] = *reinterpret_cast<const U4*>(kp + c * 32 + hi * 16 + 8);
                sc[ns] = __builtin_amdgcn_wmma_f32_16x16x32_bf16(
                    false, qf[c].v, false, kf.v, (short)0, sc[ns], false, false);
            }
        }

        // ---- scale + causal mask ----
#pragma unroll
        for (int ns = 0; ns < 2; ++ns)
#pragma unroll
            for (int r = 0; r < 8; ++r) {
                const int qrow = qw + r + hi * 8;
                const int krow = kv + ns * 16 + lr;
                sc[ns][r] = (krow <= qrow) ? sc[ns][r] * scale : -__builtin_inff();
            }

        // ---- online softmax (16-lane reductions per row) ----
#pragma unroll
        for (int r = 0; r < 8; ++r) {
            float mx = fmaxf(sc[0][r], sc[1][r]);
#pragma unroll
            for (int off = 8; off >= 1; off >>= 1)
                mx = fmaxf(mx, __shfl_xor(mx, off, 32));
            const float mnew  = fmaxf(mrow[r], mx);
            const float alpha = __expf(mrow[r] - mnew);
            const float p0 = __expf(sc[0][r] - mnew);
            const float p1 = __expf(sc[1][r] - mnew);
            sc[0][r] = p0; sc[1][r] = p1;
            float rs = p0 + p1;
#pragma unroll
            for (int off = 8; off >= 1; off >>= 1)
                rs += __shfl_xor(rs, off, 32);
            lrow[r] = lrow[r] * alpha + rs;
            mrow[r] = mnew;
#pragma unroll
            for (int dc = 0; dc < 4; ++dc) accO[dc][r] *= alpha;
        }

        // ---- P: C-layout regs -> LDS -> A-layout frag (per-wave, DS in-order) ----
#pragma unroll
        for (int ns = 0; ns < 2; ++ns)
#pragma unroll
            for (int r = 0; r < 8; ++r)
                Plds[wid][r + hi * 8][ns * 16 + lr] = (__bf16)sc[ns][r];

        FragA pf;
        const __bf16* pp = &Plds[wid][lr][0];
        pf.q[0] = *reinterpret_cast<const U4*>(pp + hi * 16);
        pf.q[1] = *reinterpret_cast<const U4*>(pp + hi * 16 + 8);

        // ---- O += P @ V : V is (Dk,S) per head -> contiguous B-frags ----
#pragma unroll
        for (int dc = 0; dc < 4; ++dc) {
            FragA vf;
            const __bf16* vp = vbase + ((size_t)(dc * 16 + lr) << 11) + kv + hi * 16;
            vf.q[0] = *reinterpret_cast<const U4*>(vp);
            vf.q[1] = *reinterpret_cast<const U4*>(vp + 8);
            accO[dc] = __builtin_amdgcn_wmma_f32_16x16x32_bf16(
                false, pf.v, false, vf.v, (short)0, accO[dc], false, false);
        }
    }

    // ---- normalize + store ctx (B, S, H*Dk) bf16 ----
#pragma unroll
    for (int dc = 0; dc < 4; ++dc)
#pragma unroll
        for (int r = 0; r < 8; ++r) {
            const int qrow = qw + r + hi * 8;
            const float o = accO[dc][r] / lrow[r];
            ctx[((size_t)(b * 2048 + qrow) * 1024) + h * 64 + dc * 16 + lr] = (__bf16)o;
        }
}

// ---------------------------------------------------------------------------
// Host-side orchestration
// ---------------------------------------------------------------------------
extern "C" void kernel_launch(void* const* d_in, const int* in_sizes, int n_in,
                              void* d_out, int out_size, void* d_ws, size_t ws_size,
                              hipStream_t stream) {
    (void)in_sizes; (void)n_in; (void)out_size; (void)ws_size;

    const float* x  = (const float*)d_in[0];
    const float* wq = (const float*)d_in[1];
    const float* bq = (const float*)d_in[2];
    const float* wk = (const float*)d_in[3];
    const float* bk = (const float*)d_in[4];
    const float* wv = (const float*)d_in[5];
    const float* bv = (const float*)d_in[6];
    const float* wo = (const float*)d_in[7];
    const float* bo = (const float*)d_in[8];
    float* out = (float*)d_out;

    const int Bb = 4, S = 2048, D = 1024, H = 16;
    const size_t tok = (size_t)Bb * S;

    char* ws = (char*)d_ws;
    __bf16* xb   = (__bf16*)ws; ws += tok * D * sizeof(__bf16);
    __bf16* wqt  = (__bf16*)ws; ws += (size_t)D * D * sizeof(__bf16);
    __bf16* wkt  = (__bf16*)ws; ws += (size_t)D * D * sizeof(__bf16);
    __bf16* wvt  = (__bf16*)ws; ws += (size_t)D * D * sizeof(__bf16);
    __bf16* wot  = (__bf16*)ws; ws += (size_t)D * D * sizeof(__bf16);
    __bf16* qb   = (__bf16*)ws; ws += tok * D * sizeof(__bf16);
    __bf16* kb   = (__bf16*)ws; ws += tok * D * sizeof(__bf16);
    __bf16* vb   = (__bf16*)ws; ws += tok * D * sizeof(__bf16);
    __bf16* ctxb = (__bf16*)ws; ws += tok * D * sizeof(__bf16);

    {
        const int nx = (int)(tok * D);
        f32_to_bf16_kernel<<<nx / 1024, 256, 0, stream>>>(x, xb, nx);
        dim3 gt(D / 32, D / 32);
        f32_to_bf16_transpose_kernel<<<gt, 256, 0, stream>>>(wq, wqt, D, D);
        f32_to_bf16_transpose_kernel<<<gt, 256, 0, stream>>>(wk, wkt, D, D);
        f32_to_bf16_transpose_kernel<<<gt, 256, 0, stream>>>(wv, wvt, D, D);
        f32_to_bf16_transpose_kernel<<<gt, 256, 0, stream>>>(wo, wot, D, D);
    }

    dim3 gg(D / 128, (unsigned)(tok / 128));   // (8, 64)
    gemm_bias_bf16_kernel<<<gg, 256, 0, stream>>>(xb, wqt, bq, qb, (int)tok, D, D, 1);
    gemm_bias_bf16_kernel<<<gg, 256, 0, stream>>>(xb, wkt, bk, kb, (int)tok, D, D, 1);
    gemm_bias_bf16_kernel<<<gg, 256, 0, stream>>>(xb, wvt, bv, vb, (int)tok, D, D, 2);

    dim3 ga(S / 64, Bb * H);                   // (32, 64)
    attn_causal_kernel<<<ga, 128, 0, stream>>>(qb, kb, vb, ctxb);

    gemm_bias_bf16_kernel<<<gg, 256, 0, stream>>>(ctxb, wot, bo, out, (int)tok, D, D, 0);
}
